// GCN1_86354612453594
// MI455X (gfx1250) — compile-verified, matrix operand order checked by reading, not run on hardware
//
#include <hip/hip_runtime.h>

// ---------------------------------------------------------------------------
// 2-layer GCN for MI455X (gfx1250, wave32, WMMA).
//  - GEMMs: v_wmma_f32_16x16x32_bf16, W staged in LDS (ds_load_b128 feeds the
//    matrix pipe). Each wave owns two 16-row strips so every B fragment feeds
//    two WMMAs; N is chunked over blockIdx.y (4 tiles/block) to keep live
//    VGPRs ~110 and avoid scratch spills.
//  - Aggregation: vectorized gather + global_atomic_add_f32; H (51MB) and the
//    accumulator both fit the 192MB L2, so scatter is L2-bound, not HBM-bound.
// ---------------------------------------------------------------------------

typedef __bf16 bf16_t;
typedef __attribute__((ext_vector_type(16))) __bf16 v16bf;
typedef __attribute__((ext_vector_type(8)))  __bf16 v8bf;
typedef __attribute__((ext_vector_type(8)))  float  v8f;
typedef __attribute__((ext_vector_type(4)))  float  v4f;
typedef __attribute__((ext_vector_type(2)))  float  v2f;

union BFrag { v16bf v; v8bf h[2]; };

template <int V> struct VecOf;
template <> struct VecOf<4> { typedef v4f T; };
template <> struct VecOf<2> { typedef v2f T; };

// ---------------- degree / normalization ----------------

__global__ void k_init_deg(float* __restrict__ deg, int n) {
  int i = blockIdx.x * blockDim.x + threadIdx.x;
  if (i < n) deg[i] = 1.0f;  // self-loop contributes 1
}

__global__ void k_edge_deg(const int* __restrict__ dst, float* __restrict__ deg, int e) {
  int i = blockIdx.x * blockDim.x + threadIdx.x;
  if (i < e) atomicAdd(&deg[dst[i]], 1.0f);
}

__global__ void k_dinv(const float* __restrict__ deg, float* __restrict__ dinv, int n) {
  int i = blockIdx.x * blockDim.x + threadIdx.x;
  if (i < n) {
    float d = deg[i];
    dinv[i] = (d > 0.0f) ? rsqrtf(d) : 0.0f;
  }
}

// ---------------- precision conversion ----------------

__global__ void k_f32_to_bf16(const float* __restrict__ a, bf16_t* __restrict__ b, long long n) {
  long long i = (long long)blockIdx.x * blockDim.x + threadIdx.x;
  if (i < n) b[i] = (bf16_t)a[i];
}

__global__ void k_relu_bf16(const float* __restrict__ a, bf16_t* __restrict__ b, long long n) {
  long long i = (long long)blockIdx.x * blockDim.x + threadIdx.x;
  if (i < n) b[i] = (bf16_t)fmaxf(a[i], 0.0f);
}

// ---------------- pack W into per-lane B-fragment order ----------------
// B (32x16, bf16): lane holds column n = lane%16, g = lane/16; element e (0..15)
// maps to K = (v<4 ? 2v : 16+2(v-4)) + g*8 + (e&1), v = e/2 (ISA 7.12.2 layout).
template <int K, int N>
__global__ void k_pack_w(const float* __restrict__ W, bf16_t* __restrict__ Wp) {
  constexpr int KT = K / 32;
  constexpr int NT = N / 16;
  int idx = blockIdx.x * blockDim.x + threadIdx.x;
  if (idx >= KT * NT * 32 * 16) return;
  int e    = idx & 15;
  int lane = (idx >> 4) & 31;
  int tile = idx >> 9;          // kt*NT + nt
  int nt   = tile % NT;
  int kt   = tile / NT;
  int v    = e >> 1;
  int g    = lane >> 4;
  int n    = lane & 15;
  int kk   = (v < 4 ? 2 * v : 16 + 2 * (v - 4)) + g * 8 + (e & 1);
  Wp[idx] = (bf16_t)W[(long long)(kt * 32 + kk) * N + nt * 16 + n];
}

// ---------------- WMMA GEMM: C[MxN] = A[MxK] (bf16) * Wp (packed bf16) ------
// W staged in LDS. One wave handles TWO 16-row strips (32 rows); blockIdx.y
// selects a 4-tile (64-column) N-chunk -> acc = 2*4*v8f = 64 VGPRs, no spills.
template <int K, int N>
__global__ void __launch_bounds__(256, 1)
k_gemm_wmma(const bf16_t* __restrict__ A,
            const bf16_t* __restrict__ Wp,
            float* __restrict__ C, int M) {
  constexpr int KT = K / 32;
  constexpr int NT = N / 16;
  constexpr int NC = 4;                  // N tiles per block (chunk)
  constexpr int TE = KT * NT * 32 * 16;  // packed bf16 elements

  __shared__ __align__(16) bf16_t lw[TE];

  const int tid = threadIdx.x;
  // cooperative 16B-per-thread fill of the W stage (64KB / 16KB)
#pragma unroll
  for (int off = 0; off < TE; off += 256 * 8) {
    int o = off + tid * 8;
    *(v8bf*)(lw + o) = *(const v8bf*)(Wp + o);
  }
  __syncthreads();

  const int lane = tid & 31;
  const int ntBase = blockIdx.y * NC;
  const long long wave = (long long)blockIdx.x * 8 + (tid >> 5);
  const int row0 = (int)(wave * 32);  // two 16-row strips
  if (row0 >= M) return;              // wave-uniform

  const int m = lane & 15;
  const int g = lane >> 4;
  int rr0 = row0 + m;       if (rr0 >= M) rr0 = M - 1;
  int rr1 = row0 + 16 + m;  if (rr1 >= M) rr1 = M - 1;
  const bf16_t* arow0 = A + (long long)rr0 * K + g * 8;
  const bf16_t* arow1 = A + (long long)rr1 * K + g * 8;

  v8f acc0[NC], acc1[NC];
#pragma unroll
  for (int t = 0; t < NC; ++t) {
    v8f z = {};
    acc0[t] = z;
    acc1[t] = z;
  }

#pragma unroll
  for (int kt = 0; kt < KT; ++kt) {
    BFrag a0, a1;
    a0.h[0] = *(const v8bf*)(arow0 + kt * 32);
    a0.h[1] = *(const v8bf*)(arow0 + kt * 32 + 16);
    a1.h[0] = *(const v8bf*)(arow1 + kt * 32);
    a1.h[1] = *(const v8bf*)(arow1 + kt * 32 + 16);
#pragma unroll
    for (int t = 0; t < NC; ++t) {
      BFrag b;
      const bf16_t* bp = lw + ((kt * NT + ntBase + t) * 32 + lane) * 16;
      b.h[0] = *(const v8bf*)(bp);
      b.h[1] = *(const v8bf*)(bp + 8);
      acc0[t] = __builtin_amdgcn_wmma_f32_16x16x32_bf16(
          false, a0.v, false, b.v, (short)0, acc0[t], false, false);
      acc1[t] = __builtin_amdgcn_wmma_f32_16x16x32_bf16(
          false, a1.v, false, b.v, (short)0, acc1[t], false, false);
    }
  }

  // D layout: lane L -> N = L%16, M = r + 8*(L/16) for VGPR r.
  const int n0 = lane & 15;
  const int mb = (lane >> 4) * 8;
#pragma unroll
  for (int t = 0; t < NC; ++t) {
    float* c0 = C + (long long)(row0 + mb) * N + (ntBase + t) * 16 + n0;
    float* c1 = C + (long long)(row0 + 16 + mb) * N + (ntBase + t) * 16 + n0;
#pragma unroll
    for (int r = 0; r < 8; ++r) {
      if (row0 + mb + r < M)      c0[(long long)r * N] = acc0[t][r];
      if (row0 + 16 + mb + r < M) c1[(long long)r * N] = acc1[t][r];
    }
  }
}

// ---------------- aggregation ----------------
// Self-loop + bias: O[n][c] = dinv[n]^2 * H[n][c] + bias[c] (also un-poisons O)
template <int C>
__global__ void k_self_init(const float* __restrict__ H, const float* __restrict__ dinv,
                            const float* __restrict__ bias, float* __restrict__ O, int n) {
  long long i = (long long)blockIdx.x * blockDim.x + threadIdx.x;
  if (i < (long long)n * C) {
    int node = (int)(i / C);
    int c    = (int)(i % C);
    float dv = dinv[node];
    O[i] = dv * dv * H[i] + bias[c];
  }
}

// One wave per edge: b128/b64 gather of H[src], then contiguous L2-resident
// global_atomic_add_f32 into the destination row.
template <int C>
__global__ void k_scatter(const int* __restrict__ src, const int* __restrict__ dst,
                          const float* __restrict__ dinv, const float* __restrict__ H,
                          float* __restrict__ O, int e) {
  constexpr int V = C / 32;
  typedef typename VecOf<V>::T VT;
  int w    = (int)((blockIdx.x * blockDim.x + threadIdx.x) >> 5);
  int lane = (int)(threadIdx.x & 31);
  if (w >= e) return;  // wave-uniform
  int s = src[w];
  int d = dst[w];
  float wt = dinv[s] * dinv[d];
  const float* hs = H + (long long)s * C + lane * V;
  float* od = O + (long long)d * C + lane * V;
  VT hv = *(const VT*)hs;
#pragma unroll
  for (int j = 0; j < V; ++j) {
    atomicAdd(&od[j], wt * hv[j]);
  }
}

// ---------------- host-side orchestration ----------------

extern "C" void kernel_launch(void* const* d_in, const int* in_sizes, int n_in,
                              void* d_out, int out_size, void* d_ws, size_t ws_size,
                              hipStream_t stream) {
  const float* x  = (const float*)d_in[0];
  const int*   ei = (const int*)d_in[1];
  const float* W1 = (const float*)d_in[2];
  const float* b1 = (const float*)d_in[3];
  const float* W2 = (const float*)d_in[4];
  const float* b2 = (const float*)d_in[5];
  float* out = (float*)d_out;

  const int IN  = 256;
  const int HID = in_sizes[3];          // 128
  const int OUT = in_sizes[5];          // 64
  const int n   = in_sizes[0] / IN;     // 100000
  const int E   = in_sizes[1] / 2;      // 1600000
  const int* src = ei;
  const int* dst = ei + E;

  // carve workspace (256B-aligned slices)
  char* p = (char*)d_ws;
  auto carve = [&](size_t bytes) {
    char* r = p;
    p += (bytes + 255) & ~(size_t)255;
    return (void*)r;
  };
  float*  deg  = (float*)carve((size_t)n * 4);
  float*  dinv = (float*)carve((size_t)n * 4);
  bf16_t* W1p  = (bf16_t*)carve((size_t)IN * HID * 2);
  bf16_t* W2p  = (bf16_t*)carve((size_t)HID * OUT * 2);
  bf16_t* xb   = (bf16_t*)carve((size_t)n * IN * 2);   // reused as hb (n*HID*2 fits)
  float*  Hf   = (float*)carve((size_t)n * HID * 4);   // reused as H2f (n*OUT*4 fits)
  float*  A1   = (float*)carve((size_t)n * HID * 4);
  bf16_t* hb   = xb;
  float*  H2f  = Hf;
  (void)ws_size; (void)n_in; (void)out_size;

  const int T = 256;
  auto blks = [](long long work, int t) { return (unsigned)((work + t - 1) / t); };

  // normalization
  k_init_deg<<<blks(n, T), T, 0, stream>>>(deg, n);
  k_edge_deg<<<blks(E, T), T, 0, stream>>>(dst, deg, E);
  k_dinv<<<blks(n, T), T, 0, stream>>>(deg, dinv, n);

  // precision prep
  k_f32_to_bf16<<<blks((long long)n * IN, T), T, 0, stream>>>(x, xb, (long long)n * IN);
  k_pack_w<256, 128><<<blks(256 * 128, T), T, 0, stream>>>(W1, W1p);
  k_pack_w<128, 64><<<blks(128 * 64, T), T, 0, stream>>>(W2, W2p);

  const long long waves = (n + 31) / 32;  // one wave per 32-row strip

  // layer 1: H = x @ W1 ; aggregate ; ReLU
  {
    dim3 grid(blks(waves * 32, T), 128 / 64);  // N chunked: 2 x 4 tiles
    k_gemm_wmma<256, 128><<<grid, T, 0, stream>>>(xb, W1p, Hf, n);
  }
  k_self_init<128><<<blks((long long)n * 128, T), T, 0, stream>>>(Hf, dinv, b1, A1, n);
  k_scatter<128><<<blks((long long)E * 32, T), T, 0, stream>>>(src, dst, dinv, Hf, A1, E);
  k_relu_bf16<<<blks((long long)n * 128, T), T, 0, stream>>>(A1, hb, (long long)n * 128);

  // layer 2: H2 = h @ W2 ; aggregate into d_out
  {
    dim3 grid(blks(waves * 32, T), 64 / 64);   // single 4-tile chunk
    k_gemm_wmma<128, 64><<<grid, T, 0, stream>>>(hb, W2p, H2f, n);
  }
  k_self_init<64><<<blks((long long)n * 64, T), T, 0, stream>>>(H2f, dinv, b2, out, n);
  k_scatter<64><<<blks((long long)E * 32, T), T, 0, stream>>>(src, dst, dinv, H2f, out, E);
}
